// BERTGraphRel_18588618457109
// MI455X (gfx1250) — compile-verified, matrix operand order checked by reading, not run on hardware
//
#include <hip/hip_runtime.h>
#include <hip/hip_bf16.h>
#include <math.h>

// Problem dims
#define S_  256
#define B_  32
#define E_  768
#define H_  256
#define NE_ 9
#define R_  5
#define G3  768    // 3*H
#define H2  512    // 2*H
#define MB  8192   // S*B

typedef __attribute__((ext_vector_type(16))) __bf16 bf16x16;
typedef __attribute__((ext_vector_type(8)))  float  f32x8;

// ---------------------------------------------------------------------------
// WMMA bf16 fragment loader (wave32, 16x16x32).
// 16-bit A-matrix 16x32 layout (ISA 7.12.2): lanes 0-15 hold rows M=0..15 with
// K={0..7,16..23}; lanes 16-31 hold the same rows with K={8..15,24..31}.
// B (32x16, KxN) mirrors this with N striped across lanes; storing the
// B operand N-major / K-contiguous makes every lane's 16 elements two
// contiguous 16-byte runs -> global_load_b128 pairs, no conversion.
// ---------------------------------------------------------------------------
static __device__ __forceinline__ bf16x16
frag_ld(const __bf16* __restrict__ base, long rs, int lane) {
  const int l = lane & 15, hi = lane >> 4;
  bf16x16 f;
#pragma unroll
  for (int e = 0; e < 16; ++e) {
    const int v = e >> 1, h = e & 1;
    const int k = ((v < 4) ? 0 : 16) + hi * 8 + (v & 3) * 2 + h;
    f[e] = base[(long)l * rs + k];
  }
  return f;
}

// ---------------------------------------------------------------------------
// f32 -> bf16 conversion (vectorized: float4 in, 8-byte bf16x4 out)
// ---------------------------------------------------------------------------
__global__ void cvt_bf16(const float* __restrict__ in, __bf16* __restrict__ out,
                         long n4) {
  const long i = (long)blockIdx.x * blockDim.x + threadIdx.x;
  if (i >= n4) return;
  const float4 v = ((const float4*)in)[i];
  out[4 * i + 0] = (__bf16)v.x;
  out[4 * i + 1] = (__bf16)v.y;
  out[4 * i + 2] = (__bf16)v.z;
  out[4 * i + 3] = (__bf16)v.w;
}

// ---------------------------------------------------------------------------
// GEMM: C[M,N] = act(A[M,K] @ W[N,K]^T + bias[N]); A,W bf16, C f32.
// Block = 128 threads (4 waves), block tile 128x64; wave tile 32x64
// (2 A-frags x 4 B-frags -> 8 WMMA per K-chunk, B-frag reused twice).
// M % 128 == 0, N % 64 == 0, K % 32 == 0 for all uses here.
// ---------------------------------------------------------------------------
__global__ void __launch_bounds__(128)
wmma_gemm(const __bf16* __restrict__ A, int lda,
          const __bf16* __restrict__ W, int ldw,
          const float* __restrict__ bias,
          float* __restrict__ C, int ldc, int K, int act) {
  const int lane = threadIdx.x & 31;
  const int wave = threadIdx.x >> 5;
  const int row0 = blockIdx.y * 128 + wave * 32;
  const int col0 = blockIdx.x * 64;
  const int l = lane & 15, hi = lane >> 4;

  f32x8 acc[2][4];
#pragma unroll
  for (int t = 0; t < 4; ++t) {
    const float bv = bias ? bias[col0 + t * 16 + l] : 0.f;
#pragma unroll
    for (int j = 0; j < 8; ++j) { acc[0][t][j] = bv; acc[1][t][j] = bv; }
  }

  for (int k0 = 0; k0 < K; k0 += 32) {
    if (k0 + 32 < K) {  // prefetch next K-chunk (global_prefetch_b8)
      __builtin_prefetch(A + (long)(row0 + l) * lda + k0 + 32, 0, 1);
      __builtin_prefetch(W + (long)(col0 + l) * ldw + k0 + 32, 0, 1);
    }
    bf16x16 a0 = frag_ld(A + (long)row0 * lda + k0, lda, lane);
    bf16x16 a1 = frag_ld(A + (long)(row0 + 16) * lda + k0, lda, lane);
#pragma unroll
    for (int t = 0; t < 4; ++t) {
      bf16x16 b = frag_ld(W + (long)(col0 + t * 16) * ldw + k0, ldw, lane);
      acc[0][t] = __builtin_amdgcn_wmma_f32_16x16x32_bf16(
          false, a0, false, b, (short)0, acc[0][t], false, false);
      acc[1][t] = __builtin_amdgcn_wmma_f32_16x16x32_bf16(
          false, a1, false, b, (short)0, acc[1][t], false, false);
    }
  }

#pragma unroll
  for (int rr = 0; rr < 2; ++rr)
#pragma unroll
    for (int t = 0; t < 4; ++t)
#pragma unroll
      for (int j = 0; j < 8; ++j) {
        float v = acc[rr][t][j];
        if (act) v = fmaxf(v, 0.f);
        C[(long)(row0 + rr * 16 + j + hi * 8) * ldc + col0 + t * 16 + l] = v;
      }
}

// ---------------------------------------------------------------------------
// Persistent GRU scan (one direction). One 1024-thread block = 32 waves.
// h kept in LDS (f32 state + bf16 WMMA A-operand copy). Whh is pre-converted
// bf16 (L2-resident, loaded fresh each of 256 steps -> halving those bytes
// and dropping per-step cvts directly shortens the serial critical path).
// Per step: m = h @ Whh^T + bhh (96 tiles, 3 per wave), then fused gates.
// Dynamic LDS: 32*256*4 + 32*256*2 + 32*768*4 = 144 KB (<320 KB/WGP).
// ---------------------------------------------------------------------------
__global__ void __launch_bounds__(1024)
gru_scan(const float* __restrict__ pre,    // (S,B,3H), includes bih
         const __bf16* __restrict__ Whh,   // (3H,H) bf16
         const float* __restrict__ bhh,    // (3H)
         float* __restrict__ out, int ldo, int colOff, int reverse) {
  extern __shared__ char smem[];
  float*  hF = (float*)smem;                                   // 32*256
  __bf16* hB = (__bf16*)(smem + (size_t)B_ * H_ * 4);          // 32*256
  float*  m  = (float*)(smem + (size_t)B_ * H_ * 4 + (size_t)B_ * H_ * 2);

  const int tid = threadIdx.x;
  const int lane = tid & 31, wave = tid >> 5;
  const int l = lane & 15, hi = lane >> 4;

  for (int i = tid; i < B_ * H_; i += 1024) { hF[i] = 0.f; hB[i] = (__bf16)0.f; }
  __syncthreads();

  for (int step = 0; step < S_; ++step) {
    const int s = reverse ? (S_ - 1 - step) : step;

    // m = hB @ Whh^T + bhh  (32 x 768)
#pragma unroll
    for (int i = 0; i < 3; ++i) {
      const int T = wave * 3 + i;        // 0..95
      const int mrow = (T & 1) * 16;
      const int col0 = (T >> 1) * 16;
      f32x8 acc;
      const float bv = bhh[col0 + l];
#pragma unroll
      for (int j = 0; j < 8; ++j) acc[j] = bv;
      for (int k0 = 0; k0 < H_; k0 += 32) {
        bf16x16 a = frag_ld(hB + mrow * H_ + k0, H_, lane);
        bf16x16 b = frag_ld(Whh + (long)col0 * H_ + k0, H_, lane);
        acc = __builtin_amdgcn_wmma_f32_16x16x32_bf16(
            false, a, false, b, (short)0, acc, false, false);
      }
#pragma unroll
      for (int j = 0; j < 8; ++j)
        m[(mrow + j + hi * 8) * G3 + col0 + l] = acc[j];
    }
    __syncthreads();

    // gates + state update
    const float* p = pre + (long)s * B_ * G3;
    for (int idx = tid; idx < B_ * H_; idx += 1024) {
      const int b = idx >> 8, k = idx & 255;
      const float* pb = p + (long)b * G3;
      const float* mb = m + (long)b * G3;
      const float r = 1.f / (1.f + expf(-(pb[k] + mb[k])));
      const float z = 1.f / (1.f + expf(-(pb[H_ + k] + mb[H_ + k])));
      const float n = tanhf(pb[2 * H_ + k] + r * mb[2 * H_ + k]);
      const float h = (1.f - z) * n + z * hF[idx];
      hF[idx] = h;
      hB[idx] = (__bf16)h;
      out[((long)s * B_ + b) * ldo + colOff + k] = h;
    }
    __syncthreads();
  }
}

// ---------------------------------------------------------------------------
// softmax over R=5 + emit bf16 adjacency in both GCN layouts:
//   adjF[b,r,s1,s2] (fw: rows=s, cols=j)   adjB[b,r,s2,s1] (bw view)
// ---------------------------------------------------------------------------
__global__ void softmax_tr(const float* __restrict__ in,
                           __bf16* __restrict__ adjF,
                           __bf16* __restrict__ adjB) {
  const long idx = (long)blockIdx.x * blockDim.x + threadIdx.x;
  const long total = (long)S_ * S_ * B_;
  if (idx >= total) return;
  const int b = (int)(idx % B_);
  const long q = idx / B_;
  const int s2 = (int)(q % S_);
  const int s1 = (int)(q / S_);
  const float* p = in + idx * R_;
  float mx = p[0];
  for (int r = 1; r < R_; ++r) mx = fmaxf(mx, p[r]);
  float e[R_], sum = 0.f;
  for (int r = 0; r < R_; ++r) { e[r] = expf(p[r] - mx); sum += e[r]; }
  const float inv = 1.f / sum;
  for (int r = 0; r < R_; ++r) {
    const __bf16 v = (__bf16)(e[r] * inv);
    adjF[(((long)b * R_ + r) * S_ + s1) * S_ + s2] = v;
    adjB[(((long)b * R_ + r) * S_ + s2) * S_ + s1] = v;
  }
}

// xp (j,b,r,k) f32  ->  xpT (b,r,k,j) bf16  (K-contiguous B-operand layout)
__global__ void xp_tr(const float* __restrict__ xp, __bf16* __restrict__ xpT) {
  const long idx = (long)blockIdx.x * blockDim.x + threadIdx.x;
  const long total = (long)MB * R_ * H_;
  if (idx >= total) return;
  const int k = (int)(idx % H_);
  long q = idx / H_;
  const int r = (int)(q % R_); q /= R_;
  const int b = (int)(q % B_);
  const int j = (int)(q / B_);
  xpT[(((long)b * R_ + r) * H_ + k) * S_ + j] = (__bf16)xp[idx];
}

// ---------------------------------------------------------------------------
// GCN aggregation: out[s,b,colOff+k] = sum_r relu( sum_j A[s,j] * Xp[j,k] ).
// adjX (b,r,s,j) and xpT (b,r,k,j) are bf16, fully contiguous fragment loads.
// Grid: (H/64, S/64, B), block 128 (4 waves, 16x64 per wave).
// ---------------------------------------------------------------------------
__global__ void __launch_bounds__(128)
gcn_apply(const __bf16* __restrict__ adjX, const __bf16* __restrict__ xpT,
          float* __restrict__ out, int colOff) {
  const int lane = threadIdx.x & 31;
  const int wave = threadIdx.x >> 5;
  const int b = blockIdx.z;
  const int row0 = blockIdx.y * 64 + wave * 16;  // s
  const int col0 = blockIdx.x * 64;              // k
  const int l = lane & 15, hi = lane >> 4;

  f32x8 sum[4];
#pragma unroll
  for (int t = 0; t < 4; ++t)
#pragma unroll
    for (int j = 0; j < 8; ++j) sum[t][j] = 0.f;

  for (int r = 0; r < R_; ++r) {
    f32x8 acc[4];
#pragma unroll
    for (int t = 0; t < 4; ++t)
#pragma unroll
      for (int j = 0; j < 8; ++j) acc[t][j] = 0.f;

    const __bf16* ab = adjX + (((long)b * R_ + r) * S_ + row0) * S_;
    const __bf16* xb = xpT + ((long)b * R_ + r) * H_ * (long)S_;
    for (int j0 = 0; j0 < S_; j0 += 32) {
      bf16x16 a = frag_ld(ab + j0, S_, lane);
#pragma unroll
      for (int t = 0; t < 4; ++t) {
        bf16x16 bf = frag_ld(xb + (long)(col0 + t * 16) * S_ + j0, S_, lane);
        acc[t] = __builtin_amdgcn_wmma_f32_16x16x32_bf16(
            false, a, false, bf, (short)0, acc[t], false, false);
      }
    }
#pragma unroll
    for (int t = 0; t < 4; ++t)
#pragma unroll
      for (int j = 0; j < 8; ++j) sum[t][j] += fmaxf(acc[t][j], 0.f);
  }

#pragma unroll
  for (int t = 0; t < 4; ++t)
#pragma unroll
    for (int j = 0; j < 8; ++j)
      out[(((long)(row0 + j + hi * 8)) * B_ + b) * H2 + colOff + col0 + t * 16 + l]
          = sum[t][j];
}

// ---------------------------------------------------------------------------
// Small-N dense layers (N = 9 or 5): plain VALU, negligible FLOPs.
// ---------------------------------------------------------------------------
__global__ void small_fc(const float* __restrict__ A, int lda, int K,
                         const float* __restrict__ W, int ldw, int wOff,
                         const float* __restrict__ bias,
                         float* __restrict__ out, int N, int total) {
  const int idx = blockIdx.x * blockDim.x + threadIdx.x;
  if (idx >= total) return;
  const int mm = idx / N, n = idx - mm * N;
  const float* a = A + (long)mm * lda;
  const float* w = W + (long)n * ldw + wOff;
  float acc = bias ? bias[n] : 0.f;
  for (int k = 0; k < K; ++k) acc += a[k] * w[k];
  out[idx] = acc;
}

__global__ void rel_combine(const float* __restrict__ a, const float* __restrict__ c,
                            const float* __restrict__ bias, float* __restrict__ out) {
  const long idx = (long)blockIdx.x * blockDim.x + threadIdx.x;
  const long total = (long)S_ * S_ * B_ * R_;
  if (idx >= total) return;
  const int r = (int)(idx % R_);
  long q = idx / R_;
  const int b = (int)(q % B_); q /= B_;
  const int s2 = (int)(q % S_);
  const int s1 = (int)(q / S_);
  out[idx] = a[((long)s1 * B_ + b) * R_ + r] + c[((long)s2 * B_ + b) * R_ + r] + bias[r];
}

__global__ void add_vec(const float* __restrict__ a, const float* __restrict__ b,
                        float* __restrict__ o, long n) {
  const long i = (long)blockIdx.x * blockDim.x + threadIdx.x;
  if (i < n) o[i] = a[i] + b[i];
}

// ---------------------------------------------------------------------------
// Host-side orchestration
// ---------------------------------------------------------------------------
static void cvt(const float* f, __bf16* b, long n, hipStream_t s) {
  const long n4 = n / 4;
  cvt_bf16<<<(int)((n4 + 255) / 256), 256, 0, s>>>(f, b, n4);
}

static void run_bigru(const __bf16* xb, int K,
                      const __bf16* Wih, const __bf16* Whh,
                      const float* bih, const float* bhh,
                      float* preF, float* preB, float* y, hipStream_t stream) {
  const dim3 grdP(G3 / 64, MB / 128);
  wmma_gemm<<<grdP, 128, 0, stream>>>(xb, K, Wih, K, bih, preF, G3, K, 0);
  wmma_gemm<<<grdP, 128, 0, stream>>>(xb, K, Wih + (long)G3 * K, K, bih + G3, preB, G3, K, 0);
  const size_t sh = (size_t)B_ * H_ * 4 + (size_t)B_ * H_ * 2 + (size_t)B_ * G3 * 4;
  gru_scan<<<1, 1024, sh, stream>>>(preF, Whh, bhh, y, H2, 0, 0);
  gru_scan<<<1, 1024, sh, stream>>>(preB, Whh + (long)G3 * H_, bhh + G3, y, H2, H_, 1);
}

static void run_pred(const __bf16* xxb,  // bf16 of block input (8192 x 512)
                     const __bf16* neWihB, const __bf16* neWhhB,
                     const float* nebih, const float* nebhh,
                     const float* fcNeW, const float* fcNeB,
                     const __bf16* t0Wb, const float* t0B,
                     const __bf16* t1Wb, const float* t1B,
                     const float* relW, const float* relB,
                     float* preF, float* preB, float* xne,
                     float* t0, float* t1, float* av, float* cv,
                     float* outNe, float* outRel, hipStream_t stream) {
  run_bigru(xxb, H2, neWihB, neWhhB, nebih, nebhh, preF, preB, xne, stream);
  small_fc<<<(MB * NE_ + 255) / 256, 256, 0, stream>>>(
      xne, H2, H2, fcNeW, H2, 0, fcNeB, outNe, NE_, MB * NE_);
  wmma_gemm<<<dim3(H_ / 64, MB / 128), 128, 0, stream>>>(xxb, H2, t0Wb, H2, t0B, t0, H_, H2, 1);
  wmma_gemm<<<dim3(H_ / 64, MB / 128), 128, 0, stream>>>(xxb, H2, t1Wb, H2, t1B, t1, H_, H2, 1);
  small_fc<<<(MB * R_ + 255) / 256, 256, 0, stream>>>(
      t0, H_, H_, relW, H2, 0, nullptr, av, R_, MB * R_);
  small_fc<<<(MB * R_ + 255) / 256, 256, 0, stream>>>(
      t1, H_, H_, relW, H2, H_, nullptr, cv, R_, MB * R_);
  const long tot = (long)S_ * S_ * B_ * R_;
  rel_combine<<<(int)((tot + 255) / 256), 256, 0, stream>>>(av, cv, relB, outRel);
}

extern "C" void kernel_launch(void* const* d_in, const int* in_sizes, int n_in,
                              void* d_out, int out_size, void* d_ws, size_t ws_size,
                              hipStream_t stream) {
  (void)in_sizes; (void)n_in; (void)out_size; (void)ws_size;

  const float* x      = (const float*)d_in[0];
  const float* eWih0  = (const float*)d_in[1];
  const float* eWhh0  = (const float*)d_in[2];
  const float* ebih0  = (const float*)d_in[3];
  const float* ebhh0  = (const float*)d_in[4];
  const float* eWih1  = (const float*)d_in[5];
  const float* eWhh1  = (const float*)d_in[6];
  const float* ebih1  = (const float*)d_in[7];
  const float* ebhh1  = (const float*)d_in[8];
  const float* fcEncW = (const float*)d_in[9];
  const float* fcEncB = (const float*)d_in[10];
  const float* neWih  = (const float*)d_in[11];
  const float* neWhh  = (const float*)d_in[12];
  const float* nebih  = (const float*)d_in[13];
  const float* nebhh  = (const float*)d_in[14];
  const float* fcNeW  = (const float*)d_in[15];
  const float* fcNeB  = (const float*)d_in[16];
  const float* t0W    = (const float*)d_in[17];
  const float* t0B    = (const float*)d_in[18];
  const float* t1W    = (const float*)d_in[19];
  const float* t1B    = (const float*)d_in[20];
  const float* relW   = (const float*)d_in[21];
  const float* relB   = (const float*)d_in[22];
  const float* gfwW   = (const float*)d_in[23];
  const float* gfwB   = (const float*)d_in[24];
  const float* gbwW   = (const float*)d_in[25];
  const float* gbwB   = (const float*)d_in[26];

  // ---- workspace carve (16B aligned) ----
  char* p = (char*)d_ws;
  auto allocF = [&](long n) { float* r = (float*)p;  p += (size_t)n * 4; return r; };
  auto allocB = [&](long n) { __bf16* r = (__bf16*)p; p += (((size_t)n * 2 + 15) / 16) * 16; return r; };

  float* preF = allocF((long)MB * G3);
  float* preB = allocF((long)MB * G3);
  float* h1   = allocF((long)MB * H2);
  float* h2   = allocF((long)MB * H2);
  float* enc  = allocF((long)MB * H2);
  float* xne  = allocF((long)MB * H2);
  float* t0   = allocF((long)MB * H_);
  float* t1   = allocF((long)MB * H_);
  float* av   = allocF((long)MB * R_);
  float* cv   = allocF((long)MB * R_);
  float* xpf  = allocF((long)MB * R_ * H_);
  float* xpb  = allocF((long)MB * R_ * H_);
  float* hcat = allocF((long)MB * H2);
  float* enc2 = allocF((long)MB * H2);

  __bf16* wih0b  = allocB((long)2 * G3 * E_);
  __bf16* wih1b  = allocB((long)2 * G3 * H2);
  __bf16* newihb = allocB((long)2 * G3 * H2);
  __bf16* whh0b  = allocB((long)2 * G3 * H_);
  __bf16* whh1b  = allocB((long)2 * G3 * H_);
  __bf16* newhhb = allocB((long)2 * G3 * H_);
  __bf16* fcencb = allocB((long)H2 * H2);
  __bf16* t0wb   = allocB((long)H_ * H2);
  __bf16* t1wb   = allocB((long)H_ * H2);
  __bf16* gfwb   = allocB((long)R_ * H_ * H2);
  __bf16* gbwb   = allocB((long)R_ * H_ * H2);
  __bf16* actb   = allocB((long)MB * E_);          // activation staging (reused)
  __bf16* adjF   = allocB((long)B_ * R_ * S_ * S_);
  __bf16* adjB   = allocB((long)B_ * R_ * S_ * S_);
  __bf16* xptf   = allocB((long)B_ * R_ * H_ * S_);
  __bf16* xptb   = allocB((long)B_ * R_ * H_ * S_);

  float* outNe1  = (float*)d_out;
  float* outRel1 = outNe1 + (long)MB * NE_;
  float* outNe2  = outRel1 + (long)S_ * S_ * B_ * R_;
  float* outRel2 = outNe2 + (long)MB * NE_;

  // ---- one-time weight conversions (all L2-resident afterwards) ----
  cvt(eWih0, wih0b, (long)2 * G3 * E_, stream);
  cvt(eWih1, wih1b, (long)2 * G3 * H2, stream);
  cvt(neWih, newihb, (long)2 * G3 * H2, stream);
  cvt(eWhh0, whh0b, (long)2 * G3 * H_, stream);
  cvt(eWhh1, whh1b, (long)2 * G3 * H_, stream);
  cvt(neWhh, newhhb, (long)2 * G3 * H_, stream);
  cvt(fcEncW, fcencb, (long)H2 * H2, stream);
  cvt(t0W, t0wb, (long)H_ * H2, stream);
  cvt(t1W, t1wb, (long)H_ * H2, stream);
  cvt(gfwW, gfwb, (long)R_ * H_ * H2, stream);
  cvt(gbwW, gbwb, (long)R_ * H_ * H2, stream);

  // ---- encoder stack ----
  cvt(x, actb, (long)MB * E_, stream);
  run_bigru(actb, E_, wih0b, whh0b, ebih0, ebhh0, preF, preB, h1, stream);

  cvt(h1, actb, (long)MB * H2, stream);
  run_bigru(actb, H2, wih1b, whh1b, ebih1, ebhh1, preF, preB, h2, stream);

  cvt(h2, actb, (long)MB * H2, stream);
  wmma_gemm<<<dim3(H2 / 64, MB / 128), 128, 0, stream>>>(
      actb, H2, fcencb, H2, fcEncB, enc, H2, H2, 0);

  // ---- prediction block 1 (actb := bf16(enc), reused by xp GEMMs too) ----
  cvt(enc, actb, (long)MB * H2, stream);
  run_pred(actb, newihb, newhhb, nebih, nebhh, fcNeW, fcNeB,
           t0wb, t0B, t1wb, t1B, relW, relB,
           preF, preB, xne, t0, t1, av, cv, outNe1, outRel1, stream);

  // ---- GCN ----
  const long nsm = (long)S_ * S_ * B_;
  softmax_tr<<<(int)((nsm + 255) / 256), 256, 0, stream>>>(outRel1, adjF, adjB);

  wmma_gemm<<<dim3((R_ * H_) / 64, MB / 128), 128, 0, stream>>>(
      actb, H2, gfwb, H2, gfwB, xpf, R_ * H_, H2, 0);
  wmma_gemm<<<dim3((R_ * H_) / 64, MB / 128), 128, 0, stream>>>(
      actb, H2, gbwb, H2, gbwB, xpb, R_ * H_, H2, 0);

  const long nxp = (long)MB * R_ * H_;
  xp_tr<<<(int)((nxp + 255) / 256), 256, 0, stream>>>(xpf, xptf);
  xp_tr<<<(int)((nxp + 255) / 256), 256, 0, stream>>>(xpb, xptb);

  gcn_apply<<<dim3(H_ / 64, S_ / 64, B_), 128, 0, stream>>>(adjF, xptf, hcat, 0);
  gcn_apply<<<dim3(H_ / 64, S_ / 64, B_), 128, 0, stream>>>(adjB, xptb, hcat, H_);

  const long ntot = (long)MB * H2;
  add_vec<<<(int)((ntot + 255) / 256), 256, 0, stream>>>(enc, hcat, enc2, ntot);

  // ---- prediction block 2 ----
  cvt(enc2, actb, (long)MB * H2, stream);
  run_pred(actb, newihb, newhhb, nebih, nebhh, fcNeW, fcNeB,
           t0wb, t0B, t1wb, t1B, relW, relB,
           preF, preB, xne, t0, t1, av, cv, outNe2, outRel2, stream);
}
